// DGCNN_seg_23278722744323
// MI455X (gfx1250) — compile-verified, matrix operand order checked by reading, not run on hardware
//
#include <hip/hip_runtime.h>
#include <hip/hip_bf16.h>
#include <stdint.h>

// ============================================================================
// DGCNN segmentation forward pass for MI455X (gfx1250, wave32, WMMA).
//
// Precision:
//  - kNN Gram matrices: exact fp32 via V_WMMA_F32_16X16X4_F32.
//  - Dense/edge GEMMs: f16 in, f32 accumulate via V_WMMA_F32_16X16X32_F16.
// Codegen (round 3):
//  - All loop-invariant path selection hoisted out of the k-loops (no per-
//    iteration SALU branch ladders).
//  - 2-stage software pipelining: A fragments / dist operands for step i+1
//    are issued before the WMMA of step i; W slabs double-buffered in LDS.
//  - No predicated loads anywhere: clamp addresses + select values.
// ============================================================================

typedef __attribute__((ext_vector_type(16))) _Float16 v16h;
typedef __attribute__((ext_vector_type(8)))  _Float16 v8h;
typedef __attribute__((ext_vector_type(8)))  float    v8f;
typedef __attribute__((ext_vector_type(4)))  float    v4f;
typedef __attribute__((ext_vector_type(2)))  float    v2f;

#define Bc   4
#define Np   2048
#define KNN  40
#define NEDGE (Bc * Np * KNN)   // 327,680 edges
#define NPTS  (Bc * Np)         // 8192 points
#define BN_EPS 1e-5f

// f16 A/B fragment K-index within a 32-wide K slab (ISA 7.12.2):
// element j of v16h -> k = (j>=8)*16 + (lane>=16)*8 + (j&7)  (contiguous runs)
__device__ __forceinline__ int frag_k(int lane, int j) {
  return ((j >> 3) << 4) | ((lane >> 4) << 3) | (j & 7);
}

// ---------------------------------------------------------------------------
// Generic f16-WMMA GEMM: out[M,Nout] = act(A[M,Kd] * W[Kd,Nout] + bias)
// EDGE=1: A row r = edge (bn = r/KNN); cols [0,Dx) = x_i, [Dx,2Dx) = x_j-x_i.
// ACT: 0=none, 1=relu, 2=leaky-relu(0.2)
// Block = 256 threads = 8 waves; one 16x16 tile per wave stacked in M.
// W slab double-buffered in LDS, column-major -> B fragment = 2x ds_load_b128.
// ---------------------------------------------------------------------------
template<int EDGE, int ACT>
__global__ __launch_bounds__(256)
void gemm_wmma(const float* __restrict__ A,   // dense A (EDGE=0)
               const float* __restrict__ X,   // [NPTS, Dx] (EDGE=1)
               const int*   __restrict__ nbr, // [NPTS, KNN] (EDGE=1)
               int Dx,
               const float* __restrict__ W,   // [Kd, Nout] row-major
               const float* __restrict__ bias,
               float* __restrict__ out,       // [M, Nout]
               int M, int Kd, int Nout)
{
  __shared__ _Float16 Bs[2][16 * 32];         // [buf][col][k]
  const int lane = threadIdx.x & 31;
  const int wid  = threadIdx.x >> 5;
  const int col0 = blockIdx.x * 16;
  const int row0 = blockIdx.y * 128 + wid * 16;
  const int rloc = lane & 15;
  const int hi   = lane >> 4;

  // Clamp the row: garbage rows only affect their own (never-stored) outputs.
  int rglob = row0 + rloc;
  if (rglob >= M) rglob = M - 1;

  int pt = 0, nb = 0;
  if (EDGE) {
    int bn = rglob / KNN;                     // b*Np + n
    pt = bn;
    nb = (bn / Np) * Np + nbr[rglob];         // b*Np + j
  }

  // Stage W[k0..k0+32) x [col0..col0+16) into Bs[buf] (column-major,
  // unconditional clamped loads + value select).
  auto stage = [&](int k0, int buf) {
#pragma unroll
    for (int it = 0; it < 2; ++it) {
      int e  = threadIdx.x + it * 256;        // 512 elements, 256 threads
      int kk = e >> 4, cc = e & 15;
      int kc = k0 + kk < Kd ? k0 + kk : Kd - 1;
      int cc_c = col0 + cc < Nout ? col0 + cc : Nout - 1;
      float w = W[(size_t)kc * Nout + cc_c];
      w = (k0 + kk < Kd && col0 + cc < Nout) ? w : 0.f;
      Bs[buf][cc * 32 + kk] = (_Float16)w;
    }
  };

  // Fast A fragment: 4x b128 loads (requires Kd%32==0; EDGE also Dx%8==0).
  auto load_a_fast = [&](int k0) -> v16h {
    v16h af;
#pragma unroll
    for (int half = 0; half < 2; ++half) {
      const int kg0 = k0 + half * 16 + hi * 8;     // 8 contiguous K values
      v4f u0, u1;
      if (EDGE) {
        const bool inA = (kg0 < Dx);
        const int  c   = inA ? kg0 : kg0 - Dx;
        const v4f* pi = (const v4f*)(X + (size_t)pt * Dx + c);
        const v4f* pj = (const v4f*)(X + (size_t)nb * Dx + c);
        v4f xi0 = pi[0], xi1 = pi[1];
        v4f xj0 = pj[0], xj1 = pj[1];
        u0 = inA ? xi0 : (xj0 - xi0);
        u1 = inA ? xi1 : (xj1 - xi1);
      } else {
        const v4f* p = (const v4f*)(A + (size_t)rglob * Kd + kg0);
        u0 = p[0]; u1 = p[1];
      }
#pragma unroll
      for (int t = 0; t < 4; ++t) {
        af[half * 8 + t]     = (_Float16)u0[t];
        af[half * 8 + 4 + t] = (_Float16)u1[t];
      }
    }
    return af;
  };

  // Slow A fragment: elementwise, clamp + select (no exec divergence).
  auto load_a_slow = [&](int k0) -> v16h {
    v16h af;
#pragma unroll
    for (int j = 0; j < 16; ++j) {
      const int kg = k0 + frag_k(lane, j);
      const int kc = kg < Kd ? kg : Kd - 1;
      float av;
      if (EDGE) {
        const bool inA = (kc < Dx);
        const int  c   = inA ? kc : kc - Dx;
        float xi = X[(size_t)pt * Dx + c];
        float xj = X[(size_t)nb * Dx + c];
        av = inA ? xi : (xj - xi);
      } else {
        av = A[(size_t)rglob * Kd + kc];
      }
      af[j] = (_Float16)(kg < Kd ? av : 0.f);
    }
    return af;
  };

  auto load_b = [&](int buf) -> v16h {
    const v8h* bp = (const v8h*)(Bs[buf] + rloc * 32 + hi * 8);
    v8h b0 = bp[0];   // k = hi*8 .. +7
    v8h b1 = bp[2];   // k = 16+hi*8 .. +7
    v16h bf;
#pragma unroll
    for (int t = 0; t < 8; ++t) { bf[t] = b0[t]; bf[8 + t] = b1[t]; }
    return bf;
  };

  v8f acc = {};
  const bool allfast = ((Kd & 31) == 0) && (!EDGE || (Dx & 7) == 0);

  stage(0, 0);
  if (allfast) {
    // Pipelined: A(i+1) + W-slab(i+1) issued before WMMA(i).
    v16h af = load_a_fast(0);
    int buf = 0;
    for (int k0 = 0; k0 < Kd; k0 += 32) {
      __syncthreads();                       // slab(k0) ready; old reads done
      const bool more = (k0 + 32 < Kd);
      if (more) stage(k0 + 32, buf ^ 1);
      v16h af_next = {};
      if (more) af_next = load_a_fast(k0 + 32);
      v16h bf = load_b(buf);
      acc = __builtin_amdgcn_wmma_f32_16x16x32_f16(false, af, false, bf,
                                                   (short)0, acc, false, false);
      af  = af_next;
      buf ^= 1;
    }
  } else {
    int buf = 0;
    for (int k0 = 0; k0 < Kd; k0 += 32) {
      __syncthreads();
      if (k0 + 32 < Kd) stage(k0 + 32, buf ^ 1);
      v16h af = load_a_slow(k0);
      v16h bf = load_b(buf);
      acc = __builtin_amdgcn_wmma_f32_16x16x32_f16(false, af, false, bf,
                                                   (short)0, acc, false, false);
      buf ^= 1;
    }
  }

  const int c = col0 + rloc;
  if (c < Nout) {
    const float bv = bias ? bias[c] : 0.f;
#pragma unroll
    for (int rr = 0; rr < 8; ++rr) {
      int m = row0 + rr + (hi << 3);          // C/D layout: M = vgpr + 8*half
      if (m < M) {
        float v = acc[rr] + bv;
        if (ACT == 1) v = v > 0.f ? v : 0.f;
        if (ACT == 2) v = v >= 0.f ? v : 0.2f * v;
        out[(size_t)m * Nout + c] = v;
      }
    }
  }
}

// ---------------------------------------------------------------------------
// Squared pairwise distance: d2[b,m,n] = |x_m|^2 + |x_n|^2 - 2 x_m.x_n
// Gram tile via exact fp32 WMMA (16x16x4). Path selection hoisted; even-D
// main loop is vectorized + software-pipelined; one masked tail step.
// ---------------------------------------------------------------------------
__global__ __launch_bounds__(128)
void rowsq_kernel(const float* __restrict__ X, float* __restrict__ sq,
                  int Ntot, int D) {
  int i = blockIdx.x * blockDim.x + threadIdx.x;
  if (i < Ntot) {
    float s = 0.f;
    for (int k = 0; k < D; ++k) { float v = X[(size_t)i * D + k]; s += v * v; }
    sq[i] = s;
  }
}

__global__ __launch_bounds__(128)
void dist_wmma(const float* __restrict__ X, const float* __restrict__ sq,
               float* __restrict__ d2, int D)
{
  const int lane = threadIdx.x & 31;
  const int wid  = threadIdx.x >> 5;
  const int n0   = blockIdx.x * 16;
  const int m0   = blockIdx.y * 64 + wid * 16;
  const int b    = blockIdx.z;
  const size_t base = (size_t)b * Np;
  const int rloc = lane & 15;
  const int hi   = lane >> 4;
  const float* rowA = X + (base + m0 + rloc) * D;
  const float* rowB = X + (base + n0 + rloc) * D;

  v8f acc = {};
  const int Dmain = D & ~3;
  if ((D & 1) == 0) {
    // 8B-aligned float2 loads, 2-stage pipeline (D=128 path).
    v2f a = {}, bb = {};
    if (Dmain > 0) {
      a  = *(const v2f*)(rowA + hi * 2);
      bb = *(const v2f*)(rowB + hi * 2);
    }
    for (int k0 = 0; k0 < Dmain; k0 += 4) {
      v2f an = {}, bn = {};
      if (k0 + 4 < Dmain) {
        an = *(const v2f*)(rowA + k0 + 4 + hi * 2);
        bn = *(const v2f*)(rowB + k0 + 4 + hi * 2);
      }
      acc = __builtin_amdgcn_wmma_f32_16x16x4_f32(false, a, false, bb,
                                                  (short)0, acc, false, false);
      a = an; bb = bn;
    }
  } else {
    for (int k0 = 0; k0 < Dmain; k0 += 4) {
      v2f a, bb;
#pragma unroll
      for (int j = 0; j < 2; ++j) {
        int k = k0 + hi * 2 + j;
        a[j]  = rowA[k];
        bb[j] = rowB[k];
      }
      acc = __builtin_amdgcn_wmma_f32_16x16x4_f32(false, a, false, bb,
                                                  (short)0, acc, false, false);
    }
  }
  if (Dmain < D) {                            // masked tail, once
    v2f a, bb;
#pragma unroll
    for (int j = 0; j < 2; ++j) {
      int k  = Dmain + hi * 2 + j;
      int kc = k < D ? k : D - 1;
      float va = rowA[kc], vb = rowB[kc];
      a[j]  = k < D ? va : 0.f;
      bb[j] = k < D ? vb : 0.f;
    }
    acc = __builtin_amdgcn_wmma_f32_16x16x4_f32(false, a, false, bb,
                                                (short)0, acc, false, false);
  }

  const int n = n0 + rloc;
#pragma unroll
  for (int rr = 0; rr < 8; ++rr) {
    int m = m0 + rr + hi * 8;
    d2[((size_t)b * Np + m) * Np + n] =
        sq[base + m] + sq[base + n] - 2.f * acc[rr];
  }
}

// One thread per (b,n): insertion top-K smallest. Stable on ties (lower column
// index first), matching jax.lax.top_k.
__global__ __launch_bounds__(128)
void topk_kernel(const float* __restrict__ d2, int* __restrict__ nbr) {
  int t = blockIdx.x * blockDim.x + threadIdx.x;   // b*Np + n
  if (t >= NPTS) return;
  const float* row = d2 + (size_t)t * Np;
  float vals[KNN]; int ids[KNN];
  for (int k = 0; k < KNN; ++k) { vals[k] = 3.4e38f; ids[k] = 0; }
  for (int m = 0; m < Np; ++m) {
    float v = row[m];
    if (v < vals[KNN - 1]) {
      int p = KNN - 1;
      while (p > 0 && vals[p - 1] > v) {
        vals[p] = vals[p - 1]; ids[p] = ids[p - 1]; --p;
      }
      vals[p] = v; ids[p] = m;
    }
  }
  for (int k = 0; k < KNN; ++k) nbr[t * KNN + k] = ids[k];
}

// Training-mode BatchNorm stats: one block per channel.
__global__ __launch_bounds__(256)
void colstats(const float* __restrict__ h, int M, int C,
              float* __restrict__ mean, float* __restrict__ rstd) {
  __shared__ float ssum[256], ssq[256];
  const int c = blockIdx.x;
  float s = 0.f, q = 0.f;
  for (int r = threadIdx.x; r < M; r += blockDim.x) {
    float v = h[(size_t)r * C + c]; s += v; q += v * v;
  }
  ssum[threadIdx.x] = s; ssq[threadIdx.x] = q;
  __syncthreads();
  for (int w = blockDim.x >> 1; w > 0; w >>= 1) {
    if (threadIdx.x < w) {
      ssum[threadIdx.x] += ssum[threadIdx.x + w];
      ssq[threadIdx.x]  += ssq[threadIdx.x + w];
    }
    __syncthreads();
  }
  if (threadIdx.x == 0) {
    float m = ssum[0] / (float)M;
    float v = ssq[0] / (float)M - m * m;
    mean[c] = m;
    rstd[c] = rsqrtf(v + BN_EPS);
  }
}

__global__ __launch_bounds__(256)
void bn_lrelu(float* __restrict__ h, size_t total, int C,
              const float* __restrict__ mean, const float* __restrict__ rstd,
              const float* __restrict__ g, const float* __restrict__ be) {
  size_t i = (size_t)blockIdx.x * blockDim.x + threadIdx.x;
  size_t stride = (size_t)gridDim.x * blockDim.x;
  for (; i < total; i += stride) {
    int c = (int)(i % C);
    float v = g[c] * (h[i] - mean[c]) * rstd[c] + be[c];
    h[i] = v >= 0.f ? v : 0.2f * v;
  }
}

// max over K neighbors: out[(b*Np+n), c] = max_k h[((b*Np+n)*KNN+k), c]
__global__ __launch_bounds__(256)
void maxpool_k(const float* __restrict__ h, float* __restrict__ out, int C) {
  int i = blockIdx.x * blockDim.x + threadIdx.x;
  if (i >= NPTS * C) return;
  int bn = i / C, c = i - bn * C;
  float m = -3.4e38f;
  for (int k = 0; k < KNN; ++k) {
    float v = h[((size_t)bn * KNN + k) * C + c];
    m = v > m ? v : m;
  }
  out[i] = m;
}

// global max over N points: out[b,c] = max_n h[(b*Np+n), c]
__global__ __launch_bounds__(256)
void maxpool_n(const float* __restrict__ h, float* __restrict__ out, int C) {
  int i = blockIdx.x * blockDim.x + threadIdx.x;
  if (i >= Bc * C) return;
  int b = i / C, c = i - b * C;
  float m = -3.4e38f;
  for (int n = 0; n < Np; ++n) {
    float v = h[((size_t)b * Np + n) * C + c];
    m = v > m ? v : m;
  }
  out[i] = m;
}

// x0 = pos @ T (per batch), xcat = [x0, feat]  -> [NPTS, 19]
__global__ __launch_bounds__(256)
void apply_transform(const float* __restrict__ pos, const float* __restrict__ feat,
                     const float* __restrict__ Tm, float* __restrict__ xcat) {
  int p = blockIdx.x * blockDim.x + threadIdx.x;
  if (p >= NPTS) return;
  int b = p / Np;
  float x = pos[p * 3], y = pos[p * 3 + 1], z = pos[p * 3 + 2];
#pragma unroll
  for (int j = 0; j < 3; ++j)
    xcat[p * 19 + j] = x * Tm[b * 9 + j] + y * Tm[b * 9 + 3 + j] + z * Tm[b * 9 + 6 + j];
  for (int c = 0; c < 16; ++c) xcat[p * 19 + 3 + c] = feat[p * 16 + c];
}

__global__ __launch_bounds__(256)
void addk(const float* __restrict__ a, const float* __restrict__ b,
          float* __restrict__ o, int n) {
  int i = blockIdx.x * blockDim.x + threadIdx.x;
  if (i < n) o[i] = a[i] + b[i];
}

// ============================================================================
// Host orchestration
// ============================================================================
extern "C" void kernel_launch(void* const* d_in, const int* in_sizes, int n_in,
                              void* d_out, int out_size, void* d_ws, size_t ws_size,
                              hipStream_t stream) {
  (void)in_sizes; (void)n_in; (void)out_size; (void)ws_size;

  // ---- input unpack --------------------------------------------------------
  // Assumed flattening: jax tree_leaves with sorted dict keys at every level:
  //   batch_indices, features,
  //   params.conv[i][l]{W,b,be,g}, params.lin[i][l]{W,b}, params.lt[i]{W,b},
  //   params.t1[l]{W,b,be,g}, params.t2[0]{W,b,be,g}, params.t3[l]{W,b},
  //   params.t4{W,b},
  //   positions
  int pi = 0;
  const int*   batch_idx = (const int*)d_in[pi++]; (void)batch_idx;
  const float* feat = (const float*)d_in[pi++];
  const float *cW[3][2], *cb[3][2], *cbe[3][2], *cg[3][2];
  for (int i = 0; i < 3; ++i)
    for (int l = 0; l < 2; ++l) {
      cW[i][l]  = (const float*)d_in[pi++];
      cb[i][l]  = (const float*)d_in[pi++];
      cbe[i][l] = (const float*)d_in[pi++];
      cg[i][l]  = (const float*)d_in[pi++];
    }
  const float *lW[3][2], *lb[3][2];
  for (int i = 0; i < 3; ++i)
    for (int l = 0; l < 2; ++l) {
      lW[i][l] = (const float*)d_in[pi++];
      lb[i][l] = (const float*)d_in[pi++];
    }
  const float *ltW[3], *ltb[3];
  for (int i = 0; i < 3; ++i) {
    ltW[i] = (const float*)d_in[pi++];
    ltb[i] = (const float*)d_in[pi++];
  }
  const float *t1W[2], *t1b[2], *t1be[2], *t1g[2];
  for (int l = 0; l < 2; ++l) {
    t1W[l]  = (const float*)d_in[pi++];
    t1b[l]  = (const float*)d_in[pi++];
    t1be[l] = (const float*)d_in[pi++];
    t1g[l]  = (const float*)d_in[pi++];
  }
  const float *t2W = (const float*)d_in[pi++], *t2b  = (const float*)d_in[pi++];
  const float *t2be = (const float*)d_in[pi++], *t2g = (const float*)d_in[pi++];
  const float *t3Wp[2], *t3bp[2];
  for (int l = 0; l < 2; ++l) {
    t3Wp[l] = (const float*)d_in[pi++];
    t3bp[l] = (const float*)d_in[pi++];
  }
  const float *t4W = (const float*)d_in[pi++], *t4b = (const float*)d_in[pi++];
  const float *pos = (const float*)d_in[pi++];

  // ---- workspace layout (~342 MB; overlapping lifetimes) ------------------
  char*  base = (char*)d_ws;
  size_t off  = 0;
  auto alloc = [&](size_t bytes) -> char* {
    char* p = base + off;
    off += (bytes + 255) & ~(size_t)255;
    return p;
  };
  float* sq     = (float*)alloc(sizeof(float) * NPTS);
  int*   nbr    = (int*)  alloc(sizeof(int)   * NPTS * KNN);
  float* meanb  = (float*)alloc(sizeof(float) * 1024);
  float* rstdb  = (float*)alloc(sizeof(float) * 1024);
  float* xcat   = (float*)alloc(sizeof(float) * NPTS * 19);
  float* xcur   = (float*)alloc(sizeof(float) * NPTS * 128);
  float* pooled = (float*)alloc(sizeof(float) * NPTS * 128);
  float* y1     = (float*)alloc(sizeof(float) * NPTS * 128);
  float* y2     = (float*)alloc(sizeof(float) * NPTS * 128);
  float* xlt    = (float*)alloc(sizeof(float) * NPTS * 128);
  float* g1024  = (float*)alloc(sizeof(float) * Bc * 1024);
  float* tb512  = (float*)alloc(sizeof(float) * Bc * 512);
  float* tb256  = (float*)alloc(sizeof(float) * Bc * 256);
  float* tmat9  = (float*)alloc(sizeof(float) * Bc * 9);
  // R1: d2 (64MB) then reused as h2 (160MB). R2: h1 (160MB), also t2h (32MB).
  float* R1 = (float*)alloc(sizeof(float) * (size_t)NEDGE * 128);
  float* R2 = (float*)alloc(sizeof(float) * (size_t)NEDGE * 128);
  float* d2  = R1;
  float* h2  = R1;
  float* h1  = R2;
  float* t2h = R2;

  // ---- launch helpers ------------------------------------------------------
  auto gemm_dense = [&](const float* A, const float* W, const float* bias,
                        float* out, int M, int Kd, int Nout, int act) {
    dim3 g((Nout + 15) / 16, (M + 127) / 128);
    if (act == 0)
      gemm_wmma<0,0><<<g,256,0,stream>>>(A,nullptr,nullptr,0,W,bias,out,M,Kd,Nout);
    else if (act == 1)
      gemm_wmma<0,1><<<g,256,0,stream>>>(A,nullptr,nullptr,0,W,bias,out,M,Kd,Nout);
    else
      gemm_wmma<0,2><<<g,256,0,stream>>>(A,nullptr,nullptr,0,W,bias,out,M,Kd,Nout);
  };
  auto gemm_edge = [&](const float* X, int Dx, const float* W, const float* bias,
                       float* out, int Nout) {
    dim3 g((Nout + 15) / 16, (NEDGE + 127) / 128);
    gemm_wmma<1,0><<<g,256,0,stream>>>(nullptr,X,nbr,Dx,W,bias,out,NEDGE,2*Dx,Nout);
  };
  auto knn = [&](const float* X, int D) {
    rowsq_kernel<<<(NPTS + 127) / 128, 128, 0, stream>>>(X, sq, NPTS, D);
    dim3 g(Np / 16, Np / 64, Bc);
    dist_wmma<<<g, 128, 0, stream>>>(X, sq, d2, D);
    topk_kernel<<<(NPTS + 127) / 128, 128, 0, stream>>>(d2, nbr);
  };
  auto bn_act = [&](float* h, int M, int C, const float* g, const float* be) {
    colstats<<<C, 256, 0, stream>>>(h, M, C, meanb, rstdb);
    bn_lrelu<<<2048, 256, 0, stream>>>(h, (size_t)M * C, C, meanb, rstdb, g, be);
  };
  auto pool_k = [&](const float* h, float* out, int C) {
    maxpool_k<<<(NPTS * C + 255) / 256, 256, 0, stream>>>(h, out, C);
  };

  // ---- t1: edge conv on positions (D=3, mlp [6 -> 64 -> 128]) -------------
  knn(pos, 3);
  gemm_edge(pos, 3, t1W[0], t1b[0], h1, 64);
  bn_act(h1, NEDGE, 64, t1g[0], t1be[0]);
  gemm_dense(h1, t1W[1], t1b[1], h2, NEDGE, 64, 128, 0);
  bn_act(h2, NEDGE, 128, t1g[1], t1be[1]);
  pool_k(h2, pooled, 128);

  // ---- t2: [NPTS,128] -> 1024, BN+lrelu, global max over N ----------------
  gemm_dense(pooled, t2W, t2b, t2h, NPTS, 128, 1024, 0);
  bn_act(t2h, NPTS, 1024, t2g, t2be);
  maxpool_n<<<(Bc * 1024 + 255) / 256, 256, 0, stream>>>(t2h, g1024, 1024);

  // ---- t3 (lrelu, no BN) + t4 -> 3x3 transform ----------------------------
  gemm_dense(g1024, t3Wp[0], t3bp[0], tb512, Bc, 1024, 512, 2);
  gemm_dense(tb512, t3Wp[1], t3bp[1], tb256, Bc, 512, 256, 2);
  gemm_dense(tb256, t4W, t4b, tmat9, Bc, 256, 9, 0);
  apply_transform<<<(NPTS + 255) / 256, 256, 0, stream>>>(pos, feat, tmat9, xcat);

  // ---- 3 stacked dynamic edge convs with residual linear transform --------
  const float* x = xcat;
  int Dx = 19;
  for (int i = 0; i < 3; ++i) {
    knn(x, Dx);                                           // d2 lives in R1
    gemm_edge(x, Dx, cW[i][0], cb[i][0], h1, 128);        // d2 dead; h2 reuses R1
    bn_act(h1, NEDGE, 128, cg[i][0], cbe[i][0]);
    gemm_dense(h1, cW[i][1], cb[i][1], h2, NEDGE, 128, 128, 0);
    bn_act(h2, NEDGE, 128, cg[i][1], cbe[i][1]);
    pool_k(h2, pooled, 128);
    gemm_dense(pooled, lW[i][0], lb[i][0], y1, NPTS, 128, 128, 1);  // relu
    gemm_dense(y1,     lW[i][1], lb[i][1], y2, NPTS, 128, 128, 0);
    gemm_dense(x, ltW[i], ltb[i], xlt, NPTS, Dx, 128, 0);
    float* xn = (i == 2) ? (float*)d_out : xcur;
    addk<<<(NPTS * 128 + 255) / 256, 256, 0, stream>>>(xlt, y2, xn, NPTS * 128);
    x = xn;
    Dx = 128;
  }
}
